// DenoisingMultiheadAttention_73349451481871
// MI455X (gfx1250) — matrix operand order repaired and checked by reading, not compiled, not run on hardware
//
#include <hip/hip_runtime.h>

// ---------------- problem constants ----------------
#define B_  4
#define H_  16
#define NT  512
#define NL  512
#define E_  64
#define P_  1024   // H_*E_

typedef __bf16 bf16;
typedef __attribute__((ext_vector_type(16))) __bf16 v16bf;
typedef __attribute__((ext_vector_type(8)))  __bf16 v8bf;
typedef __attribute__((ext_vector_type(8)))  float  v8f;

__device__ __forceinline__ v8f vzero8() {
    v8f z;
#pragma unroll
    for (int i = 0; i < 8; ++i) z[i] = 0.0f;
    return z;
}

__device__ __forceinline__ v16bf ld_frag(const bf16* lo, const bf16* hi) {
    union { v16bf v; v8bf h[2]; } u;
    u.h[0] = *(const v8bf*)lo;
    u.h[1] = *(const v8bf*)hi;
    return u.v;
}

__device__ __forceinline__ v8f wmma_bf16(v16bf a, v16bf b, v8f c) {
    return __builtin_amdgcn_wmma_f32_16x16x32_bf16(
        /*neg_a=*/false, a, /*neg_b=*/false, b,
        /*c_mod=*/(short)0, c, /*reuse_a=*/false, /*reuse_b=*/false);
}

// CDNA5 async memory->LDS copy (16 bytes per lane), tracked by ASYNCcnt.
// Low 32 bits of a generic LDS pointer are the LDS byte address.
__device__ __forceinline__ void async_copy16(void* lds_dst, const void* gsrc) {
    unsigned lds = (unsigned)(uintptr_t)lds_dst;
    asm volatile("global_load_async_to_lds_b128 %0, %1, off"
                 :: "v"(lds), "v"(gsrc) : "memory");
}
__device__ __forceinline__ void async_wait0() {
    asm volatile("s_wait_asynccnt 0x0" ::: "memory");
}

// ---------------------------------------------------------------------------
// Kernel A1: per (b,n): K = mu/bv (bf16), t23 = sum 0.5*mu^2/bv + 0.5*log(bv)
// ---------------------------------------------------------------------------
__global__ __launch_bounds__(256) void prep_kernel(
    const float* __restrict__ mu, const float* __restrict__ logvar,
    bf16* __restrict__ Kb, float* __restrict__ t23)
{
    const int bn = blockIdx.x;            // b*NL + n
    const int t  = threadIdx.x;
    const float* mrow = mu     + (size_t)bn * P_ + t * 4;
    const float* lrow = logvar + (size_t)bn * P_ + t * 4;
    float4 m4 = *(const float4*)mrow;
    float4 l4 = *(const float4*)lrow;
    float mv[4] = {m4.x, m4.y, m4.z, m4.w};
    float lv[4] = {l4.x, l4.y, l4.z, l4.w};
    bf16 ko[4];
    float acc = 0.0f;
#pragma unroll
    for (int j = 0; j < 4; ++j) {
        float var = __expf(lv[j]);
        float bv  = var + 8.0f;           // prior_var_u = sqrt(64) = 8
        ko[j] = (bf16)(mv[j] / bv);
        acc += 0.5f * mv[j] * mv[j] / bv + 0.5f * __logf(bv);
    }
    // 4 bf16 = 8 bytes
    *(uint2*)(Kb + (size_t)bn * P_ + t * 4) = *(const uint2*)ko;

    __shared__ float sred[256];
    sred[t] = acc;
    __syncthreads();
#pragma unroll
    for (int s = 128; s > 0; s >>= 1) {
        if (t < s) sred[t] += sred[t + s];
        __syncthreads();
    }
    if (t == 0) t23[bn] = sred[0];
}

// ---------------------------------------------------------------------------
// Kernel A2: V1t[b][p][n] = var/bv, V2t[b][p][n] = 8*mu/bv  (bf16, transposed)
// ---------------------------------------------------------------------------
__global__ __launch_bounds__(256) void transpose_kernel(
    const float* __restrict__ mu, const float* __restrict__ logvar,
    bf16* __restrict__ V1t, bf16* __restrict__ V2t)
{
    const int p0 = blockIdx.x * 64;
    const int n0 = blockIdx.y * 64;
    const int b  = blockIdx.z;
    __shared__ bf16 t1s[64 * 72];
    __shared__ bf16 t2s[64 * 72];
    const int t  = threadIdx.x;
    const int r  = t >> 4;            // 0..15
    const int c0 = (t & 15) * 4;      // 0..60

#pragma unroll
    for (int rr = 0; rr < 64; rr += 16) {
        int n = r + rr;
        size_t base = ((size_t)(b * NL + n0 + n)) * P_ + p0 + c0;
        float4 m4 = *(const float4*)(mu + base);
        float4 l4 = *(const float4*)(logvar + base);
        float mv[4] = {m4.x, m4.y, m4.z, m4.w};
        float lv[4] = {l4.x, l4.y, l4.z, l4.w};
#pragma unroll
        for (int j = 0; j < 4; ++j) {
            float var = __expf(lv[j]);
            float bv  = var + 8.0f;
            t1s[n * 72 + c0 + j] = (bf16)(var / bv);
            t2s[n * 72 + c0 + j] = (bf16)(8.0f * mv[j] / bv);
        }
    }
    __syncthreads();
#pragma unroll
    for (int rr = 0; rr < 64; rr += 16) {
        int p = r + rr;
        bf16 o1[4], o2[4];
#pragma unroll
        for (int j = 0; j < 4; ++j) {
            o1[j] = t1s[(c0 + j) * 72 + p];
            o2[j] = t2s[(c0 + j) * 72 + p];
        }
        size_t base = ((size_t)(b * P_ + p0 + p)) * NL + n0 + c0;
        *(uint2*)(V1t + base) = *(const uint2*)o1;
        *(uint2*)(V2t + base) = *(const uint2*)o2;
    }
}

// ---------------------------------------------------------------------------
// Kernel B: bias[bh][n] = masked ? -inf : log(max(pi,tiny)) - t23[b][n]
// ---------------------------------------------------------------------------
__global__ __launch_bounds__(512) void bias_kernel(
    const float* __restrict__ pi, const unsigned char* __restrict__ mask,
    const float* __restrict__ t23, float* __restrict__ biasg)
{
    const int bh = blockIdx.x;
    const int b  = bh >> 4;
    const int n  = threadIdx.x;
    float piv = pi[b * NL + n];
    bool m = (mask[bh * NL + n] != 0) || (piv <= 0.0f);
    float t = t23[b * NL + n];
    biasg[bh * NL + n] = m ? -INFINITY
                           : (__logf(fmaxf(piv, 1.1754944e-38f)) - t);
}

// ---------------------------------------------------------------------------
// Kernel C: per (bh, 64-row tile): scores -> softmax -> values -> projection
// ---------------------------------------------------------------------------
__global__ __launch_bounds__(256) void attn_kernel(
    const float* __restrict__ u,  const float* __restrict__ pb,
    const bf16*  __restrict__ Kb, const bf16*  __restrict__ V1t,
    const bf16*  __restrict__ V2t, const float* __restrict__ biasg,
    const float* __restrict__ wv, const float* __restrict__ bvg,
    float* __restrict__ out, float* __restrict__ attn)
{
    // ---- LDS layout ----
    __shared__ __align__(16) bf16 Ab[64 * 520];             // attn bf16, padded
    __shared__ __align__(16) unsigned char stage[76800];    // union staging
    __shared__ float red[64 * 4];
    __shared__ float rowmax[64];
    __shared__ float rowsum[64];
    __shared__ float pbv[64];
    __shared__ float biasv[512];

    bf16* Qs    = (bf16*)stage;              // [64][40]  phase 1
    bf16* Ks    = (bf16*)(stage + 5120);     // [512][40] phase 1
    bf16* V1s   = (bf16*)stage;              // [32][520] phase 2
    bf16* V2s   = (bf16*)(stage + 33280);    // [32][520] phase 2
    bf16* wvs   = (bf16*)(stage + 66560);    // [64][40]  phase 2
    bf16* outps = (bf16*)(stage + 71680);    // [64][40]  phase 2
    bf16* bvs   = (bf16*)stage;              // [64][40]  phase 3

    const int tid  = threadIdx.x;
    const int lane = tid & 31;
    const int wid  = tid >> 5;
    const int lm   = lane & 15;
    const int ho   = (lane >> 4) << 3;        // 0 or 8 (C-frag row offset)
    const int kb   = (lane < 16) ? 0 : 8;     // A-frag K base
    const int bb   = (lane < 16) ? 0 : 8;     // B-frag K base

    const int bh = blockIdx.y;
    const int b  = bh >> 4, h = bh & 15;
    const int m0 = blockIdx.x * 64;

    const float* ublk = u + (size_t)bh * NT * P_ + (size_t)m0 * P_;

    // preload per-row / per-col bias
    if (tid < 64) pbv[tid] = pb[bh * NT + m0 + tid];
    biasv[tid]       = biasg[bh * NL + tid];
    biasv[tid + 256] = biasg[bh * NL + tid + 256];

    // ================= Phase 1: S = Q @ K^T =================
    const int wm = wid >> 2;   // 0..1 -> rows wm*32
    const int wn = wid & 3;    // 0..3 -> cols wn*128
    v8f acc[2][8];
#pragma unroll
    for (int rt = 0; rt < 2; ++rt)
#pragma unroll
        for (int ct = 0; ct < 8; ++ct) acc[rt][ct] = vzero8();

    for (int kc = 0; kc < P_; kc += 32) {
        __syncthreads();
        {   // stage K: 512x32 bf16, async memory->LDS (no VGPR data path)
            int n0 = (tid >> 2) * 8;
            int c0 = (tid & 3) * 8;
#pragma unroll
            for (int j = 0; j < 8; ++j) {
                async_copy16(Ks + (n0 + j) * 40 + c0,
                             Kb + (size_t)(b * NL + n0 + j) * P_ + kc + c0);
            }
        }
        {   // stage Q: 64x32 f32 -> bf16 (converting copy, VGPR path)
            int r  = tid >> 2;
            int c0 = (tid & 3) * 8;
            const float* src = ublk + (size_t)r * P_ + kc + c0;
            float4 f0 = *(const float4*)src;
            float4 f1 = *(const float4*)(src + 4);
            bf16* dst = Qs + r * 40 + c0;
            dst[0] = (bf16)f0.x; dst[1] = (bf16)f0.y;
            dst[2] = (bf16)f0.z; dst[3] = (bf16)f0.w;
            dst[4] = (bf16)f1.x; dst[5] = (bf16)f1.y;
            dst[6] = (bf16)f1.z; dst[7] = (bf16)f1.w;
        }
        async_wait0();
        __syncthreads();

        v16bf afr[2];
#pragma unroll
        for (int rt = 0; rt < 2; ++rt) {
            const bf16* ar = Qs + (wm * 32 + rt * 16 + lm) * 40;
            afr[rt] = ld_frag(ar + kb, ar + kb + 16);
        }
#pragma unroll
        for (int ct = 0; ct < 8; ++ct) {
            const bf16* br = Ks + (wn * 128 + ct * 16 + lm) * 40;
            v16bf bfr = ld_frag(br + bb, br + bb + 16);
            acc[0][ct] = wmma_bf16(afr[0], bfr, acc[0][ct]);
            acc[1][ct] = wmma_bf16(afr[1], bfr, acc[1][ct]);
        }
    }
    __syncthreads();

    // ---- add projected_b + bias, softmax over n ----
    float rowb[2][8];
#pragma unroll
    for (int rt = 0; rt < 2; ++rt)
#pragma unroll
        for (int v = 0; v < 8; ++v)
            rowb[rt][v] = pbv[wm * 32 + rt * 16 + v + ho];

#pragma unroll
    for (int rt = 0; rt < 2; ++rt)
#pragma unroll
        for (int ct = 0; ct < 8; ++ct) {
            float cb = biasv[wn * 128 + ct * 16 + lm];
#pragma unroll
            for (int v = 0; v < 8; ++v) acc[rt][ct][v] += rowb[rt][v] + cb;
        }

    // per-row max (wave-local 128 cols, then cross-wave via LDS)
#pragma unroll
    for (int rt = 0; rt < 2; ++rt) {
        float mx[8];
#pragma unroll
        for (int v = 0; v < 8; ++v) {
            float m = acc[rt][0][v];
#pragma unroll
            for (int ct = 1; ct < 8; ++ct) m = fmaxf(m, acc[rt][ct][v]);
            m = fmaxf(m, __shfl_xor(m, 1, 32));
            m = fmaxf(m, __shfl_xor(m, 2, 32));
            m = fmaxf(m, __shfl_xor(m, 4, 32));
            m = fmaxf(m, __shfl_xor(m, 8, 32));
            mx[v] = m;
        }
        if (lm == 0) {
#pragma unroll
            for (int v = 0; v < 8; ++v)
                red[(wm * 32 + rt * 16 + v + ho) * 4 + wn] = mx[v];
        }
    }
    __syncthreads();
    if (tid < 64)
        rowmax[tid] = fmaxf(fmaxf(red[tid * 4 + 0], red[tid * 4 + 1]),
                            fmaxf(red[tid * 4 + 2], red[tid * 4 + 3]));
    __syncthreads();

    // exp + per-row sum
#pragma unroll
    for (int rt = 0; rt < 2; ++rt) {
        float rm[8], sm[8];
#pragma unroll
        for (int v = 0; v < 8; ++v) {
            rm[v] = rowmax[wm * 32 + rt * 16 + v + ho];
            sm[v] = 0.0f;
        }
#pragma unroll
        for (int ct = 0; ct < 8; ++ct)
#pragma unroll
            for (int v = 0; v < 8; ++v) {
                float e = __expf(acc[rt][ct][v] - rm[v]);
                acc[rt][ct][v] = e;
                sm[v] += e;
            }
#pragma unroll
        for (int v = 0; v < 8; ++v) {
            float s = sm[v];
            s += __shfl_xor(s, 1, 32);
            s += __shfl_xor(s, 2, 32);
            s += __shfl_xor(s, 4, 32);
            s += __shfl_xor(s, 8, 32);
            sm[v] = s;
        }
        if (lm == 0) {
#pragma unroll
            for (int v = 0; v < 8; ++v)
                red[(wm * 32 + rt * 16 + v + ho) * 4 + wn] = sm[v];
        }
    }
    __syncthreads();
    if (tid < 64)
        rowsum[tid] = red[tid * 4 + 0] + red[tid * 4 + 1] +
                      red[tid * 4 + 2] + red[tid * 4 + 3];
    __syncthreads();

    // normalize: write attn f32 to d_out and bf16 to LDS
    float* attn_blk = attn + (size_t)bh * NT * NL + (size_t)m0 * NL;
#pragma unroll
    for (int rt = 0; rt < 2; ++rt) {
        float inv[8];
#pragma unroll
        for (int v = 0; v < 8; ++v)
            inv[v] = 1.0f / rowsum[wm * 32 + rt * 16 + v + ho];
#pragma unroll
        for (int ct = 0; ct < 8; ++ct) {
            int n = wn * 128 + ct * 16 + lm;
#pragma unroll
            for (int v = 0; v < 8; ++v) {
                int row = wm * 32 + rt * 16 + v + ho;
                float a = acc[rt][ct][v] * inv[v];
                attn_blk[(size_t)row * NL + n] = a;
                Ab[row * 520 + n] = (bf16)a;
            }
        }
    }
    __syncthreads();

    // ================= Phase 2: out_p = (A@V1)*u + A@V2, acc_e += out_p @ wv^T
    const int mt  = wid & 3;          // row tile 0..3
    const int pt  = wid >> 2;         // p col tile 0..1 (GEMM2)
    const int ee0 = (wid >> 2) * 2;   // e col tiles {ee0, ee0+1} (GEMM3)
    v8f acce[2];
    acce[0] = vzero8();
    acce[1] = vzero8();

    for (int pc = 0; pc < P_; pc += 32) {
        __syncthreads();
        {   // stage V1s,V2s : [32][512] bf16, async memory->LDS
            int pl = tid >> 3;
            int c0 = (tid & 7) * 64;
            size_t gbase = (size_t)(b * P_ + pc + pl) * NL + c0;
#pragma unroll
            for (int j = 0; j < 64; j += 8) {
                async_copy16(V1s + pl * 520 + c0 + j, V1t + gbase + j);
                async_copy16(V2s + pl * 520 + c0 + j, V2t + gbase + j);
            }
        }
        {   // stage wvs[e][p_local] bf16 from w_v f32 [h][e][p]
            int e  = tid >> 2;
            int c0 = (tid & 3) * 8;
            const float* src = wv + (size_t)h * E_ * P_ + (size_t)e * P_ + pc + c0;
            float4 f0 = *(const float4*)src;
            float4 f1 = *(const float4*)(src + 4);
            bf16* dst = wvs + e * 40 + c0;
            dst[0] = (bf16)f0.x; dst[1] = (bf16)f0.y;
            dst[2] = (bf16)f0.z; dst[3] = (bf16)f0.w;
            dst[4] = (bf16)f1.x; dst[5] = (bf16)f1.y;
            dst[6] = (bf16)f1.z; dst[7] = (bf16)f1.w;
        }
        async_wait0();
        __syncthreads();

        // GEMM2: X1/X2 tile (16x16) per wave, contraction over n=512
        v8f x1 = vzero8(), x2 = vzero8();
        const bf16* arow = Ab + (mt * 16 + lm) * 520;
        const bf16* b1r  = V1s + (pt * 16 + lm) * 520;
        const bf16* b2r  = V2s + (pt * 16 + lm) * 520;
#pragma unroll
        for (int ks = 0; ks < NL; ks += 32) {
            v16bf a  = ld_frag(arow + ks + kb, arow + ks + kb + 16);
            v16bf b1 = ld_frag(b1r + ks + bb, b1r + ks + bb + 16);
            v16bf b2 = ld_frag(b2r + ks + bb, b2r + ks + bb + 16);
            x1 = wmma_bf16(a, b1, x1);
            x2 = wmma_bf16(a, b2, x2);
        }
        // elementwise with projected_u, round to bf16 into LDS
#pragma unroll
        for (int v = 0; v < 8; ++v) {
            int m = mt * 16 + v + ho;
            int p = pc + pt * 16 + lm;
            float uu = ublk[(size_t)m * P_ + p];
            float op = x1[v] * uu + x2[v];
            outps[m * 40 + pt * 16 + lm] = (bf16)op;
        }
        __syncthreads();

        // GEMM3: acc_e += out_p(64x32) @ wv^T(32x64)  (single K-step)
        {
            const bf16* ar = outps + (mt * 16 + lm) * 40;
            v16bf a = ld_frag(ar + kb, ar + kb + 16);
#pragma unroll
            for (int j = 0; j < 2; ++j) {
                const bf16* br = wvs + ((ee0 + j) * 16 + lm) * 40;
                v16bf bf_ = ld_frag(br + bb, br + bb + 16);
                acce[j] = wmma_bf16(a, bf_, acce[j]);
            }
        }
    }

    // ================= Phase 3: acc_e += A(64x512) @ b_v^T(512x64) ==========
    for (int nc = 0; nc < NL; nc += 32) {
        __syncthreads();
        {   // stage bvs[e][n_local] bf16 from b_v f32 [h][e][n]
            int e  = tid >> 2;
            int c0 = (tid & 3) * 8;
            const float* src = bvg + (size_t)h * E_ * NL + (size_t)e * NL + nc + c0;
            float4 f0 = *(const float4*)src;
            float4 f1 = *(const float4*)(src + 4);
            bf16* dst = bvs + e * 40 + c0;
            dst[0] = (bf16)f0.x; dst[1] = (bf16)f0.y;
            dst[2] = (bf16)f0.z; dst[3] = (bf16)f0.w;
            dst[4] = (bf16)f1.x; dst[5] = (bf16)f1.y;
            dst[6] = (bf16)f1.z; dst[7] = (bf16)f1.w;
        }
        __syncthreads();
        const bf16* ar = Ab + (mt * 16 + lm) * 520;
        v16bf a = ld_frag(ar + nc + kb, ar + nc + kb + 16);
#pragma unroll
        for (int j = 0; j < 2; ++j) {
            const bf16* br = bvs + ((ee0 + j) * 16 + lm) * 40;
            v16bf bf_ = ld_frag(br + bb, br + bb + 16);
            acce[j] = wmma_bf16(a, bf_, acce[j]);
        }
    }

    // write out [bh][m][e]
    float* ob = out + (size_t)bh * NT * E_ + (size_t)m0 * E_;
#pragma unroll
    for (int j = 0; j < 2; ++j)
#pragma unroll
        for (int v = 0; v < 8; ++v) {
            int m = mt * 16 + v + ho;
            int e = (ee0 + j) * 16 + lm;
            ob[(size_t)m * E_ + e] = acce[j][v];
        }
}

// ---------------------------------------------------------------------------
extern "C" void kernel_launch(void* const* d_in, const int* in_sizes, int n_in,
                              void* d_out, int out_size, void* d_ws, size_t ws_size,
                              hipStream_t stream) {
    (void)in_sizes; (void)n_in; (void)out_size; (void)ws_size;

    const float*         u    = (const float*)d_in[0];
    const float*         pb   = (const float*)d_in[1];
    const float*         mu   = (const float*)d_in[2];
    const float*         lv   = (const float*)d_in[3];
    const float*         pi   = (const float*)d_in[4];
    const float*         wv   = (const float*)d_in[5];
    const float*         bvg  = (const float*)d_in[6];
    const unsigned char* mask = (const unsigned char*)d_in[7];

    char* ws = (char*)d_ws;
    bf16*  Kb    = (bf16*)(ws);                        // 4 MiB
    bf16*  V1t   = (bf16*)(ws + 4u * 1024 * 1024);     // 4 MiB
    bf16*  V2t   = (bf16*)(ws + 8u * 1024 * 1024);     // 4 MiB
    float* t23   = (float*)(ws + 12u * 1024 * 1024);   // 8 KiB
    float* biasg = (float*)(ws + 12u * 1024 * 1024 + 8192); // 128 KiB

    float* out  = (float*)d_out;
    float* attn = out + (size_t)B_ * H_ * NT * E_;

    prep_kernel<<<dim3(B_ * NL), dim3(256), 0, stream>>>(mu, lv, Kb, t23);
    transpose_kernel<<<dim3(P_ / 64, NL / 64, B_), dim3(256), 0, stream>>>(mu, lv, V1t, V2t);
    bias_kernel<<<dim3(B_ * H_), dim3(NL), 0, stream>>>(pi, mask, t23, biasg);
    attn_kernel<<<dim3(NT / 64, B_ * H_), dim3(256), 0, stream>>>(
        u, pb, Kb, V1t, V2t, biasg, wv, bvg, out, attn);
}
